// EncoderLayer_2491081032078
// MI455X (gfx1250) — compile-verified, matrix operand order checked by reading, not run on hardware
//
#include <hip/hip_runtime.h>
#include <math.h>

typedef _Float16 v8h  __attribute__((ext_vector_type(8)));
typedef _Float16 v16h __attribute__((ext_vector_type(16)));
typedef float    v8f  __attribute__((ext_vector_type(8)));

#define TOKENS 65536   // B*N = 8*8192
#define CDIM   512
#define FDIM   2048
#define WS     64
#define NHEAD  8
#define DHEAD  64

// ---------------------------------------------------------------------------
// Fragment loader: CDNA5 16-bit A/B layout (ISA 7.12.2).
// lane l<16  : row l,    K = {0..7} U {16..23}
// lane l>=16 : row l-16, K = {8..15} U {24..31}
// Two 16-byte LDS reads per fragment. 'stride' in halves; rows padded so
// consecutive-row b128 reads hit distinct bank groups.
// ---------------------------------------------------------------------------
__device__ __forceinline__ v16h ldsFrag(const _Float16* base, int row, int stride,
                                        int kOff, int lane) {
  const int half = (lane >> 4) & 1;
  const _Float16* p = base + row * stride + kOff + half * 8;
  v8h lo = *(const v8h*)(p);
  v8h hi = *(const v8h*)(p + 16);
  return __builtin_shufflevector(lo, hi, 0,1,2,3,4,5,6,7,8,9,10,11,12,13,14,15);
}

__device__ __forceinline__ v8f wmma16(v16h a, v16h b, v8f c) {
  // D = A(16x32,f16) * B(32x16,f16) + C(16x16,f32)
  return __builtin_amdgcn_wmma_f32_16x16x32_f16(false, a, false, b, (short)0, c,
                                                false, false);
}

// ---------------------------------------------------------------------------
// Generic WMMA GEMM: out[M,N] = A16[M,K] @ Bt16[N,K]^T + bias (+epilogue)
// Block 256 threads = 8 waves (wave32). Tile 128x128, BK=32.
// Wave grid 2x4: each wave owns 64 rows x 32 cols = 4x2 tiles of 16x16.
// Double-buffered LDS + register prefetch: global loads for tile k+1 are in
// flight while tile k's 8 WMMAs execute; one barrier per k-step.
// ---------------------------------------------------------------------------
enum { EPI_F16 = 0, EPI_RES_F32 = 1, EPI_GELU_F16 = 2 };

template<int EPI>
__global__ __launch_bounds__(256)
void gemm_wmma_kernel(const _Float16* __restrict__ A, const _Float16* __restrict__ Bt,
                      const float* __restrict__ bias, const float* __restrict__ res,
                      _Float16* __restrict__ outH, float* __restrict__ outF,
                      int M, int N, int K)
{
  __shared__ _Float16 As[2][128 * 40];   // 128 rows x 32 halves, padded to 40
  __shared__ _Float16 Bs[2][128 * 40];
  const int t    = threadIdx.x;
  const int lane = t & 31;
  const int wid  = t >> 5;
  const int waveM = (wid >> 2) * 64;  // 0 / 64
  const int waveN = (wid & 3)  * 32;  // 0 / 32 / 64 / 96
  const int mBase = blockIdx.y * 128;
  const int nBase = blockIdx.x * 128;

  // Each thread stages two 16B segments of A and of B per k-step.
  const int s0 = t, s1 = t + 256;               // segment ids in [0,512)
  const int rA0 = s0 >> 2, cA0 = (s0 & 3) * 8;  // row, half-offset
  const int rA1 = s1 >> 2, cA1 = (s1 & 3) * 8;
  const size_t gA0 = (size_t)(mBase + rA0) * K + cA0;
  const size_t gA1 = (size_t)(mBase + rA1) * K + cA1;
  const size_t gB0 = (size_t)(nBase + rA0) * K + cA0;
  const size_t gB1 = (size_t)(nBase + rA1) * K + cA1;
  const int lA0 = rA0 * 40 + cA0;
  const int lA1 = rA1 * 40 + cA1;

  v8h ra0, ra1, rb0, rb1;
  auto loadTiles = [&](int kb) {
    ra0 = *(const v8h*)(A  + gA0 + kb);
    ra1 = *(const v8h*)(A  + gA1 + kb);
    rb0 = *(const v8h*)(Bt + gB0 + kb);
    rb1 = *(const v8h*)(Bt + gB1 + kb);
  };
  auto storeTiles = [&](int buf) {
    *(v8h*)&As[buf][lA0] = ra0;
    *(v8h*)&As[buf][lA1] = ra1;
    *(v8h*)&Bs[buf][lA0] = rb0;
    *(v8h*)&Bs[buf][lA1] = rb1;
  };

  v8f acc[4][2];
  #pragma unroll
  for (int i = 0; i < 4; ++i)
    #pragma unroll
    for (int j = 0; j < 2; ++j)
      acc[i][j] = (v8f){0.f,0.f,0.f,0.f,0.f,0.f,0.f,0.f};

  loadTiles(0);
  storeTiles(0);
  __syncthreads();

  for (int kb = 0; kb < K; kb += 32) {
    const int  cur  = (kb >> 5) & 1;
    const bool more = (kb + 32) < K;
    if (more) loadTiles(kb + 32);          // prefetch next tile (in flight)

    v16h af[4], bf[2];
    #pragma unroll
    for (int mt = 0; mt < 4; ++mt)
      af[mt] = ldsFrag(As[cur], waveM + mt*16 + (lane & 15), 40, 0, lane);
    #pragma unroll
    for (int nt = 0; nt < 2; ++nt)
      bf[nt] = ldsFrag(Bs[cur], waveN + nt*16 + (lane & 15), 40, 0, lane);
    #pragma unroll
    for (int mt = 0; mt < 4; ++mt)
      #pragma unroll
      for (int nt = 0; nt < 2; ++nt)
        acc[mt][nt] = wmma16(af[mt], bf[nt], acc[mt][nt]);

    if (more) {
      storeTiles(cur ^ 1);                 // deposit prefetched tile
      __syncthreads();                     // single barrier per k-step
    }
  }

  // Epilogue. C/D layout: lane&15 = column; VGPR j = row j (+8 for lanes 16-31).
  #pragma unroll
  for (int nt = 0; nt < 2; ++nt) {
    const int col = nBase + waveN + nt*16 + (lane & 15);
    const float bv = bias[col];
    #pragma unroll
    for (int mt = 0; mt < 4; ++mt) {
      const int row0 = mBase + waveM + mt*16 + ((lane >> 4) & 1) * 8;
      #pragma unroll
      for (int j = 0; j < 8; ++j) {
        const size_t idx = (size_t)(row0 + j) * N + col;
        const float val = acc[mt][nt][j] + bv;
        if (EPI == EPI_F16) {
          outH[idx] = (_Float16)val;
        } else if (EPI == EPI_RES_F32) {
          outF[idx] = val + res[idx];
        } else { // exact GELU: 0.5*x*(1+erf(x/sqrt(2)))
          const float g = 0.5f * val * (1.0f + erff(val * 0.70710678118654752f));
          outH[idx] = (_Float16)g;
        }
      }
    }
  }
}

// ---------------------------------------------------------------------------
// Windowed MHSA: one block per (window, head). 128 threads = 4 waves.
// Each wave computes 16 rows of the 64x64 attention matrix.
// ---------------------------------------------------------------------------
__global__ __launch_bounds__(128)
void attn_wmma_kernel(const _Float16* __restrict__ q, const _Float16* __restrict__ k,
                      const _Float16* __restrict__ v, _Float16* __restrict__ o)
{
  __shared__ _Float16 Qs[64 * 72];
  __shared__ _Float16 Ks[64 * 72];
  __shared__ _Float16 Vt[64 * 72];      // V transposed: Vt[d][j]
  __shared__ _Float16 Ps[4][16 * 72];   // per-wave P scratch
  const int w    = blockIdx.x;
  const int h    = blockIdx.y;
  const int t    = threadIdx.x;
  const int lane = t & 31;
  const int wid  = t >> 5;
  const size_t rowOff = (size_t)w * WS * CDIM + (size_t)h * DHEAD;

  // Stage Q, K rows (64 x 64 halves each = 512 x 16B segments)
  for (int s = t; s < 512; s += 128) {
    const int r = s >> 3, seg = s & 7;
    *(v8h*)&Qs[r*72 + seg*8] = *(const v8h*)(q + rowOff + (size_t)r*CDIM + seg*8);
    *(v8h*)&Ks[r*72 + seg*8] = *(const v8h*)(k + rowOff + (size_t)r*CDIM + seg*8);
  }
  // Stage V transposed (coalesced 16B global reads, scalar LDS scatter)
  for (int s = t; s < 512; s += 128) {
    const int j = s >> 3, seg = s & 7;
    const v8h tmp = *(const v8h*)(v + rowOff + (size_t)j*CDIM + seg*8);
    #pragma unroll
    for (int e = 0; e < 8; ++e) Vt[(seg*8 + e)*72 + j] = tmp[e];
  }
  __syncthreads();

  // S = Q K^T for row strip [16*wid, 16*wid+16); K rows serve directly as B frags
  v8f s4[4];
  #pragma unroll
  for (int i = 0; i < 4; ++i) s4[i] = (v8f){0.f,0.f,0.f,0.f,0.f,0.f,0.f,0.f};
  #pragma unroll
  for (int kb = 0; kb < 2; ++kb) {
    const v16h a = ldsFrag(Qs, 16*wid + (lane & 15), 72, kb*32, lane);
    #pragma unroll
    for (int nt = 0; nt < 4; ++nt) {
      const v16h b = ldsFrag(Ks, nt*16 + (lane & 15), 72, kb*32, lane);
      s4[nt] = wmma16(a, b, s4[nt]);
    }
  }

  // Softmax per row. Acc VGPR j = row (16*wid + j + 8*(lane>=16)), col = lane&15.
  // Row reduction = in-lane max/sum over 4 n-tiles + xor-shuffle over 16 lanes
  // (masks 1,2,4,8 keep the two half-wave row groups independent).
  _Float16* Pw = &Ps[wid][0];
  const float scale = 0.125f;  // 1/sqrt(64)
  #pragma unroll
  for (int j = 0; j < 8; ++j) {
    float x0 = s4[0][j]*scale, x1 = s4[1][j]*scale;
    float x2 = s4[2][j]*scale, x3 = s4[3][j]*scale;
    float mx = fmaxf(fmaxf(x0, x1), fmaxf(x2, x3));
    #pragma unroll
    for (int mk = 1; mk < 16; mk <<= 1) mx = fmaxf(mx, __shfl_xor(mx, mk, 32));
    const float p0 = __expf(x0 - mx), p1 = __expf(x1 - mx);
    const float p2 = __expf(x2 - mx), p3 = __expf(x3 - mx);
    float sum = p0 + p1 + p2 + p3;
    #pragma unroll
    for (int mk = 1; mk < 16; mk <<= 1) sum += __shfl_xor(sum, mk, 32);
    const float inv = 1.0f / sum;
    const int r = j + ((lane >> 4) & 1) * 8;
    const int c = lane & 15;
    Pw[r*72 +  0 + c] = (_Float16)(p0 * inv);
    Pw[r*72 + 16 + c] = (_Float16)(p1 * inv);
    Pw[r*72 + 32 + c] = (_Float16)(p2 * inv);
    Pw[r*72 + 48 + c] = (_Float16)(p3 * inv);
  }

  // O = P @ V ; B-fragments come from Vt rows (i.e. V columns)
  v8f oa[4];
  #pragma unroll
  for (int i = 0; i < 4; ++i) oa[i] = (v8f){0.f,0.f,0.f,0.f,0.f,0.f,0.f,0.f};
  #pragma unroll
  for (int kb = 0; kb < 2; ++kb) {
    const v16h a = ldsFrag(Pw, (lane & 15), 72, kb*32, lane);
    #pragma unroll
    for (int dt = 0; dt < 4; ++dt) {
      const v16h b = ldsFrag(Vt, dt*16 + (lane & 15), 72, kb*32, lane);
      oa[dt] = wmma16(a, b, oa[dt]);
    }
  }
  #pragma unroll
  for (int dt = 0; dt < 4; ++dt) {
    const int col = h*DHEAD + dt*16 + (lane & 15);
    const int r0  = 16*wid + ((lane >> 4) & 1) * 8;
    #pragma unroll
    for (int j = 0; j < 8; ++j)
      o[(size_t)(w*WS + r0 + j)*CDIM + col] = (_Float16)oa[dt][j];
  }
}

// ---------------------------------------------------------------------------
// LayerNorm over C=512, one block (128 threads) per row. Population variance.
// ---------------------------------------------------------------------------
template<bool OUT16>
__global__ __launch_bounds__(128)
void layernorm_kernel(const float* __restrict__ in, const float* __restrict__ gamma,
                      const float* __restrict__ beta, float* __restrict__ outF,
                      _Float16* __restrict__ outH)
{
  __shared__ float red[8];
  const int row  = blockIdx.x;
  const int t    = threadIdx.x;
  const int lane = t & 31, wid = t >> 5;
  const float* x = in + (size_t)row * CDIM;
  float xv[4], sum = 0.f, ssq = 0.f;
  #pragma unroll
  for (int i = 0; i < 4; ++i) { const float a = x[t + 128*i]; xv[i] = a; sum += a; ssq += a*a; }
  #pragma unroll
  for (int mk = 1; mk < 32; mk <<= 1) { sum += __shfl_xor(sum, mk, 32); ssq += __shfl_xor(ssq, mk, 32); }
  if (lane == 0) { red[wid] = sum; red[4 + wid] = ssq; }
  __syncthreads();
  sum = red[0] + red[1] + red[2] + red[3];
  ssq = red[4] + red[5] + red[6] + red[7];
  const float mean = sum * (1.0f / CDIM);
  const float var  = ssq * (1.0f / CDIM) - mean * mean;
  const float inv  = rsqrtf(var + 1e-12f);
  #pragma unroll
  for (int i = 0; i < 4; ++i) {
    const int c = t + 128*i;
    const float y = gamma[c] * (xv[i] - mean) * inv + beta[c];
    outF[(size_t)row*CDIM + c] = y;
    if (OUT16) outH[(size_t)row*CDIM + c] = (_Float16)y;
  }
}

// ---------------------------------------------------------------------------
// Converters
// ---------------------------------------------------------------------------
__global__ void f32_to_f16_kernel(const float* __restrict__ x, _Float16* __restrict__ y,
                                  size_t n) {
  const size_t i = (size_t)blockIdx.x * blockDim.x + threadIdx.x;
  if (i < n) y[i] = (_Float16)x[i];
}

// w[K][N] (row-major) -> wt[N][K] f16 (so B-tiles load like A-tiles)
__global__ void transpose_f32_to_f16_kernel(const float* __restrict__ w,
                                            _Float16* __restrict__ wt, int K, int N) {
  const size_t i = (size_t)blockIdx.x * blockDim.x + threadIdx.x;
  if (i < (size_t)K * N) {
    const int kk = (int)(i / N), nn = (int)(i % N);
    wt[(size_t)nn * K + kk] = (_Float16)w[i];
  }
}

// ---------------------------------------------------------------------------
extern "C" void kernel_launch(void* const* d_in, const int* in_sizes, int n_in,
                              void* d_out, int out_size, void* d_ws, size_t ws_size,
                              hipStream_t stream)
{
  (void)in_sizes; (void)n_in; (void)out_size; (void)ws_size;
  const float* x      = (const float*)d_in[0];
  const float* wq     = (const float*)d_in[1];
  const float* bq     = (const float*)d_in[2];
  const float* wk     = (const float*)d_in[3];
  const float* bk     = (const float*)d_in[4];
  const float* wv     = (const float*)d_in[5];
  const float* bv     = (const float*)d_in[6];
  const float* wo     = (const float*)d_in[7];
  const float* bo     = (const float*)d_in[8];
  const float* gamma1 = (const float*)d_in[9];
  const float* beta1  = (const float*)d_in[10];
  const float* w_fc1  = (const float*)d_in[11];
  const float* b_fc1  = (const float*)d_in[12];
  const float* w_fc2  = (const float*)d_in[13];
  const float* b_fc2  = (const float*)d_in[14];
  const float* gamma2 = (const float*)d_in[15];
  const float* beta2  = (const float*)d_in[16];
  float* out = (float*)d_out;

  char* wsp = (char*)d_ws;
  size_t off = 0;
  auto take = [&](size_t bytes) -> char* {
    char* p = wsp + off;
    off = (off + bytes + 255) & ~(size_t)255;
    return p;
  };

  _Float16* wqT = (_Float16*)take((size_t)CDIM * CDIM * 2);
  _Float16* wkT = (_Float16*)take((size_t)CDIM * CDIM * 2);
  _Float16* wvT = (_Float16*)take((size_t)CDIM * CDIM * 2);
  _Float16* woT = (_Float16*)take((size_t)CDIM * CDIM * 2);
  _Float16* w1T = (_Float16*)take((size_t)CDIM * FDIM * 2);
  _Float16* w2T = (_Float16*)take((size_t)FDIM * CDIM * 2);
  _Float16* xh  = (_Float16*)take((size_t)TOKENS * CDIM * 2);
  _Float16* qB  = (_Float16*)take((size_t)TOKENS * CDIM * 2);
  _Float16* kB  = (_Float16*)take((size_t)TOKENS * CDIM * 2);
  _Float16* vB  = (_Float16*)take((size_t)TOKENS * CDIM * 2);
  _Float16* aoB = (_Float16*)take((size_t)TOKENS * CDIM * 2);
  _Float16* h16 = (_Float16*)take((size_t)TOKENS * CDIM * 2);
  float*    r32 = (float*)   take((size_t)TOKENS * CDIM * 4);
  _Float16* ff  = (_Float16*)take((size_t)TOKENS * FDIM * 2);

  // 1) Precision conversion / weight transposition
  const size_t nX = (size_t)TOKENS * CDIM;
  f32_to_f16_kernel<<<dim3((unsigned)((nX + 255) / 256)), dim3(256), 0, stream>>>(x, xh, nX);
  transpose_f32_to_f16_kernel<<<dim3((CDIM*CDIM + 255) / 256), dim3(256), 0, stream>>>(wq, wqT, CDIM, CDIM);
  transpose_f32_to_f16_kernel<<<dim3((CDIM*CDIM + 255) / 256), dim3(256), 0, stream>>>(wk, wkT, CDIM, CDIM);
  transpose_f32_to_f16_kernel<<<dim3((CDIM*CDIM + 255) / 256), dim3(256), 0, stream>>>(wv, wvT, CDIM, CDIM);
  transpose_f32_to_f16_kernel<<<dim3((CDIM*CDIM + 255) / 256), dim3(256), 0, stream>>>(wo, woT, CDIM, CDIM);
  transpose_f32_to_f16_kernel<<<dim3((CDIM*FDIM + 255) / 256), dim3(256), 0, stream>>>(w_fc1, w1T, CDIM, FDIM);
  transpose_f32_to_f16_kernel<<<dim3((FDIM*CDIM + 255) / 256), dim3(256), 0, stream>>>(w_fc2, w2T, FDIM, CDIM);

  const dim3 blk(256);
  const dim3 gC(CDIM / 128, TOKENS / 128);   // (4, 512)
  const dim3 gF(FDIM / 128, TOKENS / 128);   // (16, 512)

  // 2) Q, K, V projections (bias fused, f16 out)
  gemm_wmma_kernel<EPI_F16><<<gC, blk, 0, stream>>>(xh, wqT, bq, nullptr, qB, nullptr, TOKENS, CDIM, CDIM);
  gemm_wmma_kernel<EPI_F16><<<gC, blk, 0, stream>>>(xh, wkT, bk, nullptr, kB, nullptr, TOKENS, CDIM, CDIM);
  gemm_wmma_kernel<EPI_F16><<<gC, blk, 0, stream>>>(xh, wvT, bv, nullptr, vB, nullptr, TOKENS, CDIM, CDIM);

  // 3) Windowed multi-head attention
  attn_wmma_kernel<<<dim3(TOKENS / WS, NHEAD), dim3(128), 0, stream>>>(qB, kB, vB, aoB);

  // 4) Output projection + shortcut (x, f32), then LN1 (in-place f32 + f16 copy)
  gemm_wmma_kernel<EPI_RES_F32><<<gC, blk, 0, stream>>>(aoB, woT, bo, x, nullptr, r32, TOKENS, CDIM, CDIM);
  layernorm_kernel<true><<<dim3(TOKENS), dim3(128), 0, stream>>>(r32, gamma1, beta1, r32, h16);

  // 5) MLP: fc1 + exact GELU (f16 out), fc2 + residual h (f32, in-place), LN2 -> out
  gemm_wmma_kernel<EPI_GELU_F16><<<gF, blk, 0, stream>>>(h16, w1T, b_fc1, nullptr, ff, nullptr, TOKENS, FDIM, CDIM);
  gemm_wmma_kernel<EPI_RES_F32><<<gC, blk, 0, stream>>>(ff, w2T, b_fc2, r32, nullptr, r32, TOKENS, CDIM, FDIM);
  layernorm_kernel<false><<<dim3(TOKENS), dim3(128), 0, stream>>>(r32, gamma2, beta2, out, nullptr);
}
